// SPELL_63178968924565
// MI455X (gfx1250) — matrix operand order repaired and checked by reading, not compiled, not run on hardware
//
#include <hip/hip_runtime.h>
#include <hip/hip_bf16.h>
#include <math.h>

// ---------------------------------------------------------------------------
// MI455X (gfx1250) implementation.
// Dominant cost: edge-MLP GEMMs (~750 GFLOP) -> v_wmma_f32_16x16x32_bf16.
// Memory traffic ~0.6 GB/call vs 23.3 TB/s -> compute bound, bf16 WMMA path.
// GEMM: 128 threads / 4 waves in a 2x2 grid, 32x32 per wave, 64x64 per block,
// double-buffered LDS tiles (one barrier per K-step; staging overlaps WMMA),
// all staging via b128 vector traffic with run-granular guards.
// Banded graph (dst = src +/- 1..10) -> deterministic direct band sums for
// SAGE aggregation; edge_conv segment_max via CAS float atomic-max
// (order-independent => deterministic). Workspace ~210 MB.
// ---------------------------------------------------------------------------

typedef __bf16 bf16;
typedef __bf16 v16bf __attribute__((ext_vector_type(16)));
typedef __bf16 v8bf  __attribute__((ext_vector_type(8)));
typedef float  v8f   __attribute__((ext_vector_type(8)));

#define BGR   2000   // number of graphs (GEMM M for graph-level mats)
#define CIN   1536
#define DAGG  1200
#define CH0   1024
#define FD    512
#define KCAT  2736   // DAGG + CIN
#define KPAD  2752   // KCAT padded to multiple of 32
#define LSTR  48     // LDS tile stride (elements): 96 B, 16B-aligned runs

enum { EPI_BF16 = 0, EPI_RELU_BF16 = 1, EPI_GNRELU_BF16 = 2,
       EPI_F32 = 3, EPI_F32ADD = 4, EPI_AMAX = 5 };
enum { AM_PLAIN = 0, AM_EDGE = 1 };

__device__ inline void atomicMaxF(float* addr, float val) {
  unsigned int old = *(volatile unsigned int*)addr;
  while (val > __uint_as_float(old)) {
    unsigned int assumed = old;
    old = atomicCAS((unsigned int*)addr, assumed, __float_as_uint(val));
    if (old == assumed) break;
  }
}

// ---------------------------------------------------------------------------
// Templated bf16 WMMA GEMM: C(MxN) = A(MxK) @ B(KxN), f32 accumulate.
// Block = 128 threads (4 waves, 2x2 grid, 32x32 per wave), tile 64x64,
// K staged in steps of 32 via double-buffered LDS.
// AMODE: 0 = plain row-major bf16 A (lda >= K, rows 16B aligned, data beyond
//            KrealB must be zero in memory);
//        1 = edge features [h[dst], h[src]-h[dst]] gathered from H (K=2048).
// KrealB guards the B (weight) matrix only. N must be a multiple of 16.
// ---------------------------------------------------------------------------
template<int AMODE, int EPI>
__global__ __launch_bounds__(128) void k_gemm(
    const bf16* __restrict__ A, int lda,
    const bf16* __restrict__ H,
    const int* __restrict__ gsrc, const int* __restrict__ gdst,
    const bf16* __restrict__ Bw, int ldb,
    int M, int N, int K, int KrealB,
    float* __restrict__ outF, bf16* __restrict__ outB, int ldo,
    const float* __restrict__ bias,
    const float* __restrict__ gvec, const float* __restrict__ bevec,
    const int* __restrict__ eattr, int brMode)
{
  __shared__ bf16 As[2][64][LSTR];   // [buf][m][k]
  __shared__ bf16 Bs[2][64][LSTR];   // [buf][n][k] (transposed during staging)
  const int tid = threadIdx.x, lane = tid & 31, wave = tid >> 5;
  const int mBase = blockIdx.y * 64, nBase = blockIdx.x * 64;

  // A staging: thread -> (row 0..63, k-half 0/16)
  const int sr = tid >> 1;
  const int sh = (tid & 1) << 4;
  const int aRow = mBase + sr;
  const bool rowOK = aRow < M;
  const bf16* aPtr = nullptr; int ed = 0, es = 0;
  if (AMODE == AM_PLAIN) { if (rowOK) aPtr = A + (size_t)aRow * lda; }
  else                   { if (rowOK) { ed = gdst[aRow]; es = gsrc[aRow]; } }

  // B staging: thread -> (k 0..31, 16-col run)
  const int bk = tid >> 2;
  const int bn = (tid & 3) << 4;
  const int bcol = nBase + bn;
  const bool bColOK = bcol < N;   // N multiple of 16 => run-granular guard

  auto stage = [&](int buf, int kb) {
    // ---- A tile (64x32): two b128 loads + two b128 LDS stores ----
    v8bf a0 = {}, a1 = {};
    if (rowOK) {
      if (AMODE == AM_PLAIN) {
        const v8bf* p = (const v8bf*)(aPtr + kb + sh);
        a0 = p[0]; a1 = p[1];
      } else {
        int kk = kb + sh;           // 16-aligned: run never straddles CH0
        if (kk < CH0) {
          const v8bf* p = (const v8bf*)(H + (size_t)ed * CH0 + kk);
          a0 = p[0]; a1 = p[1];
        } else {
          int k2 = kk - CH0;
          const v8bf* ps = (const v8bf*)(H + (size_t)es * CH0 + k2);
          const v8bf* pd = (const v8bf*)(H + (size_t)ed * CH0 + k2);
          v8bf s0 = ps[0], s1 = ps[1], d0 = pd[0], d1 = pd[1];
          #pragma unroll
          for (int e = 0; e < 8; ++e) {
            a0[e] = (bf16)((float)s0[e] - (float)d0[e]);
            a1[e] = (bf16)((float)s1[e] - (float)d1[e]);
          }
        }
      }
    }
    *(v8bf*)&As[buf][sr][sh]     = a0;
    *(v8bf*)&As[buf][sr][sh + 8] = a1;
    // ---- B tile (32x64): coalesced b128 loads, b16 transpose scatter ----
    int kk = kb + bk;
    v8bf b0 = {}, b1 = {};
    if (bColOK && kk < KrealB) {
      const v8bf* p = (const v8bf*)(Bw + (size_t)kk * ldb + bcol);
      b0 = p[0]; b1 = p[1];
    }
    #pragma unroll
    for (int e = 0; e < 8; ++e) {
      Bs[buf][bn + e][bk]     = b0[e];
      Bs[buf][bn + 8 + e][bk] = b1[e];
    }
    // prefetch the K block after next (gfx1250 global_prefetch_b8 path)
    if (kb + 32 < K) {
      if (AMODE == AM_PLAIN && rowOK) __builtin_prefetch(aPtr + kb + 32 + sh, 0, 1);
      if (bColOK && kk + 32 < KrealB)
        __builtin_prefetch(&Bw[(size_t)(kk + 32) * ldb + bcol], 0, 1);
    }
  };

  const int half = lane >> 4, l15 = lane & 15;
  const int wm = wave & 1, wn = wave >> 1;   // 2x2 wave grid
  v8f acc[4] = {};

  stage(0, 0);
  __syncthreads();
  for (int kb = 0; kb < K; kb += 32) {
    const int cur = (kb >> 5) & 1;
    if (kb + 32 < K) stage(cur ^ 1, kb + 32);   // overlaps with WMMA below

    // ---- fragments (all loaded up front -> one dscnt wait, 4 WMMA burst) ----
    v16bf af[2];
    #pragma unroll
    for (int mi = 0; mi < 2; ++mi) {
      const int am = (wm << 5) + (mi << 4) + l15;
      // ISA 16-bit A 16x32 layout: lanes 0-15 -> K {0..7,16..23}; lanes 16-31 -> +8
      #pragma unroll
      for (int e = 0; e < 16; ++e) {
        int k = (e & 7) + (half << 3) + ((e & 8) << 1);
        af[mi][e] = As[cur][am][k];
      }
    }
    v16bf bfr[2];
    #pragma unroll
    for (int ni = 0; ni < 2; ++ni) {
      const int bcn = (wn << 5) + (ni << 4) + l15;
      // B 32x16 layout (matches A): lanes 0-15 -> K {0..7,16..23}; lanes 16-31 -> +8
      #pragma unroll
      for (int e = 0; e < 16; ++e) { int k = (e & 7) + (half << 3) + ((e & 8) << 1); bfr[ni][e] = Bs[cur][bcn][k]; }
    }
    #pragma unroll
    for (int mi = 0; mi < 2; ++mi)
      #pragma unroll
      for (int ni = 0; ni < 2; ++ni)
        acc[mi * 2 + ni] = __builtin_amdgcn_wmma_f32_16x16x32_bf16(
            false, af[mi], false, bfr[ni], (short)0, acc[mi * 2 + ni], false, false);
    __syncthreads();
  }

  // ---- epilogue: C/D layout: VGPR r -> M = r + (half?8:0), N = lane&15 ----
  const int rowW = mBase + (wm << 5) + (half << 3);
  #pragma unroll
  for (int mi = 0; mi < 2; ++mi) {
    #pragma unroll
    for (int ni = 0; ni < 2; ++ni) {
      const int col = nBase + (wn << 5) + (ni << 4) + l15;
      if (col >= N) continue;
      #pragma unroll
      for (int r = 0; r < 8; ++r) {
        const int row = rowW + (mi << 4) + r;
        if (row >= M) continue;
        float v = acc[mi * 2 + ni][r];
        if (EPI == EPI_BF16) {
          v += bias[col];
          outB[(size_t)row * ldo + col] = (bf16)v;
        } else if (EPI == EPI_RELU_BF16) {
          v = fmaxf(v + bias[col], 0.f);
          outB[(size_t)row * ldo + col] = (bf16)v;
        } else if (EPI == EPI_GNRELU_BF16) {
          v = fmaxf(gvec[col] * (v + bias[col]) + bevec[col], 0.f);
          outB[(size_t)row * ldo + col] = (bf16)v;
        } else if (EPI == EPI_F32) {
          if (bias) v += bias[col];
          outF[(size_t)row * ldo + col] = v;
        } else if (EPI == EPI_F32ADD) {
          if (bias) v += bias[col];
          outF[(size_t)row * ldo + col] += v;
        } else if (EPI == EPI_AMAX) {
          int at = eattr[row];
          bool ok = (brMode == 2) || (brMode == 0 ? (at <= 0) : (at >= 0));
          if (ok) {
            v += bias[col];
            atomicMaxF(&outF[(size_t)gdst[row] * ldo + col], v);
          }
        }
      }
    }
  }
}

// ---------------------------------------------------------------------------
// Support kernels
// ---------------------------------------------------------------------------
__global__ void k_cvt_bf16(const float* __restrict__ s, bf16* __restrict__ d, size_t n) {
  size_t i = (size_t)blockIdx.x * 256 + threadIdx.x;
  if (i < n) d[i] = (bf16)s[i];
}
__global__ void k_fill_f32(float* p, float v, int n) {
  int i = blockIdx.x * 256 + threadIdx.x;
  if (i < n) p[i] = v;
}
__global__ void k_zero_i32(int* p, int n) {
  int i = blockIdx.x * 256 + threadIdx.x;
  if (i < n) p[i] = 0;
}
__global__ void k_count(const int* __restrict__ batch, int* __restrict__ counts, int n) {
  int i = blockIdx.x * 256 + threadIdx.x;
  if (i < n) atomicAdd(&counts[batch[i]], 1);
}
__global__ void k_scan(const int* __restrict__ counts, int* __restrict__ starts, int n) {
  if (blockIdx.x == 0 && threadIdx.x == 0) {
    int s = 0;
    for (int i = 0; i < n; ++i) { starts[i] = s; s += counts[i]; }
  }
}
// per-graph sums: mean over first 25 (padded L) and mean over all nodes
__global__ void k_gsum(const float* __restrict__ x, const int* __restrict__ starts,
                       const int* __restrict__ counts,
                       bf16* __restrict__ meanL, bf16* __restrict__ hcat) {
  int g = blockIdx.x, tid = threadIdx.x;
  int st = starts[g], cnt = counts[g];
  float invc = 1.f / fmaxf((float)cnt, 1.f);
  for (int c = tid; c < CIN; c += 256) {
    float sf = 0.f, sp = 0.f;
    for (int j = 0; j < cnt; ++j) {
      float v = x[(size_t)(st + j) * CIN + c];
      sf += v;
      if (j < 25) sp += v;   // pos >= L dropped in reference padded scatter
    }
    meanL[(size_t)g * CIN + c] = (bf16)(sp * (1.f / 25.f));
    hcat[(size_t)g * KPAD + DAGG + c] = (bf16)(sf * invc);
  }
  if (tid < (KPAD - KCAT)) hcat[(size_t)g * KPAD + KCAT + tid] = (bf16)0.f;
}
// x -> relu(g*x' + be) where x' = (finite ? x : 0); writes f32 + bf16
__global__ void k_gbr(const float* __restrict__ in, float* __restrict__ outF,
                      bf16* __restrict__ outB, const float* __restrict__ g,
                      const float* __restrict__ be, int n, int Cmask) {
  int i = blockIdx.x * 256 + threadIdx.x;
  if (i >= n) return;
  float v = in[i];
  if (v < -3.0e38f || !(v == v)) v = 0.f;  // -FLT_MAX sentinel / NaN -> 0
  int c = i & Cmask;
  v = fmaxf(g[c] * v + be[c], 0.f);
  outF[i] = v;
  outB[i] = (bf16)v;
}
// banded neighbor mean: s[d][c] = mean over allowed src in d+/-1..10 of z[src][c]
__global__ void k_band_agg(const float* __restrict__ zin, bf16* __restrict__ outbf,
                           int C, int brMode, int Bn) {
  int d = blockIdx.x;
  int c = blockIdx.y * 256 + threadIdx.x;
  if (c >= C) return;
  float s = 0.f; int cnt = 0;
  for (int off = 1; off <= 10; ++off) {
    if (brMode != 1) { int sv = d + off; if (sv < Bn)  { s += zin[(size_t)sv * C + c]; ++cnt; } }
    if (brMode != 0) { int sv = d - off; if (sv >= 0)  { s += zin[(size_t)sv * C + c]; ++cnt; } }
  }
  outbf[(size_t)d * C + c] = (bf16)(s / fmaxf((float)cnt, 1.f));
}
// write output slot 0 and channel-major xr0 (= out.T)
__global__ void k_out0(const float* __restrict__ out_acc, float* __restrict__ dout,
                       float* __restrict__ xr, int T) {
  int idx = blockIdx.x * 256 + threadIdx.x;
  if (idx >= T * FD) return;
  int t = idx / FD, c = idx - t * FD;
  float v = out_acc[idx];
  dout[idx] = v;
  xr[(size_t)c * T + t] = v;
}
// softmax over 512 channels per position
__global__ void k_softmax(const float* __restrict__ xr, float* __restrict__ sm, int T) {
  int t = blockIdx.x, tid = threadIdx.x;   // block 128
  __shared__ float red[128];
  float mx = -3.4e38f;
  for (int c = tid; c < FD; c += 128) mx = fmaxf(mx, xr[(size_t)c * T + t]);
  red[tid] = mx; __syncthreads();
  for (int s = 64; s > 0; s >>= 1) { if (tid < s) red[tid] = fmaxf(red[tid], red[tid + s]); __syncthreads(); }
  mx = red[0]; __syncthreads();
  float sum = 0.f;
  for (int c = tid; c < FD; c += 128) sum += expf(xr[(size_t)c * T + t] - mx);
  red[tid] = sum; __syncthreads();
  for (int s = 64; s > 0; s >>= 1) { if (tid < s) red[tid] += red[tid + s]; __syncthreads(); }
  float inv = 1.f / red[0];
  for (int c = tid; c < FD; c += 128) sm[(size_t)c * T + t] = expf(xr[(size_t)c * T + t] - mx) * inv;
}
// 1x1 conv 512 -> 64
__global__ void k_convin(const float* __restrict__ sm, const float* __restrict__ w_in,
                         const float* __restrict__ b_in, float* __restrict__ f, int T) {
  int t = blockIdx.x, o = threadIdx.x;  // block 64
  float acc = b_in[o];
  const float* w = w_in + (size_t)o * FD;
  for (int c = 0; c < FD; ++c) acc += w[c] * sm[(size_t)c * T + t];
  f[(size_t)o * T + t] = acc;
}
// dilated 3-tap 64->64 conv + relu + 1x1 conv, residual (ping-pong buffers)
__global__ void k_dil(const float* __restrict__ fin, float* __restrict__ fout,
                      const float* __restrict__ wd, const float* __restrict__ bd,
                      const float* __restrict__ w1, const float* __restrict__ b1,
                      int dil, int T) {
  int t = blockIdx.x, o = threadIdx.x;  // block 64
  __shared__ float hs[64];
  float acc = bd[o];
  int tm = t - dil, tp = t + dil;
  for (int i = 0; i < 64; ++i) {
    const float* w = wd + (size_t)o * 192 + i * 3;
    float a = (tm >= 0) ? fin[(size_t)i * T + tm] : 0.f;
    float b = fin[(size_t)i * T + t];
    float c = (tp < T)  ? fin[(size_t)i * T + tp] : 0.f;
    acc += w[0] * a + w[1] * b + w[2] * c;
  }
  hs[o] = fmaxf(acc, 0.f);
  __syncthreads();
  float a2 = b1[o];
  const float* w1o = w1 + o * 64;
  for (int i = 0; i < 64; ++i) a2 += w1o[i] * hs[i];
  fout[(size_t)o * T + t] = fin[(size_t)o * T + t] + a2;
}
// 1x1 conv 64 -> 512; writes channel-major xrn + transposed output slot
__global__ void k_convout(const float* __restrict__ fin, const float* __restrict__ w_out,
                          const float* __restrict__ b_out, float* __restrict__ xrn,
                          float* __restrict__ dslot, int T) {
  int t = blockIdx.x, tid = threadIdx.x;  // block 128
  __shared__ float fs[64];
  if (tid < 64) fs[tid] = fin[(size_t)tid * T + t];
  __syncthreads();
  for (int c = tid; c < FD; c += 128) {
    float acc = b_out[c];
    const float* w = w_out + (size_t)c * 64;
    for (int i = 0; i < 64; ++i) acc += w[i] * fs[i];
    xrn[(size_t)c * T + t] = acc;
    dslot[(size_t)t * FD + c] = acc;
  }
}

// ---------------------------------------------------------------------------
extern "C" void kernel_launch(void* const* d_in, const int* in_sizes, int n_in,
                              void* d_out, int out_size, void* d_ws, size_t ws_size,
                              hipStream_t stream) {
  const float* x     = (const float*)d_in[0];
  const int*   batch = (const int*)d_in[1];
  const int*   esrc  = (const int*)d_in[2];
  const int*   edst  = (const int*)d_in[3];
  const int*   eattr = (const int*)d_in[4];
  const int E  = in_sizes[2];
  const int NN = in_sizes[0] / CIN;
  float* dout = (float*)d_out;

  // params flattened as jax pytree (dict keys ASCII-sorted, lists in order):
  // 0:W011 1:W_agg 2:Wl21 3:Wr21 4:b011 5:b_agg 6:be01 7:be21 8:bl21
  // 9+e*6: ec[e]{Wa,Wb,ba,bb,be,g}  27:g01 28:g21
  // 29+r*44: refs[r]{b_in,b_out, layers 10x{b1,bd,w1,wd}, w_in(+42), w_out(+43)}
  // 161+s*3: sage3[s]{Wl,Wr,bl}
  auto P = [&](int i) -> const float* { return (const float*)d_in[5 + i]; };

  // ---- workspace layout ----
  char* wsb = (char*)d_ws;
  size_t off = 0;
  auto alloc = [&](size_t bytes) -> void* {
    off = (off + 255) & ~(size_t)255;
    void* p = wsb + off; off += bytes; return p;
  };
  bf16* wb_W011 = (bf16*)alloc((size_t)KCAT * CH0 * 2);
  bf16* wb_Wagg = (bf16*)alloc((size_t)CIN * DAGG * 2);
  bf16* wb_Wl21 = (bf16*)alloc((size_t)CH0 * CH0 * 2);
  bf16* wb_Wr21 = (bf16*)alloc((size_t)CH0 * CH0 * 2);
  bf16 *wb_Wa[3], *wb_Wb[3], *wb_sWl[3], *wb_sWr[3];
  for (int e = 0; e < 3; ++e) {
    wb_Wa[e] = (bf16*)alloc((size_t)2 * CH0 * CH0 * 2);
    wb_Wb[e] = (bf16*)alloc((size_t)CH0 * CH0 * 2);
  }
  for (int s = 0; s < 3; ++s) {
    wb_sWl[s] = (bf16*)alloc((size_t)CH0 * FD * 2);
    wb_sWr[s] = (bf16*)alloc((size_t)CH0 * FD * 2);
  }
  int*   counts = (int*)alloc((size_t)BGR * 4);
  int*   starts = (int*)alloc((size_t)BGR * 4);
  bf16*  meanL  = (bf16*)alloc((size_t)BGR * CIN * 2);
  bf16*  hcat   = (bf16*)alloc((size_t)BGR * KPAD * 2);
  bf16*  h_bf   = (bf16*)alloc((size_t)BGR * CH0 * 2);
  bf16*  m1     = (bf16*)alloc((size_t)E * CH0 * 2);
  float* z      = (float*)alloc((size_t)BGR * CH0 * 4);
  bf16*  z_bf   = (bf16*)alloc((size_t)BGR * CH0 * 2);
  float* t1     = (float*)alloc((size_t)BGR * CH0 * 4);
  float* z2     = (float*)alloc((size_t)BGR * CH0 * 4);
  bf16*  z2_bf  = (bf16*)alloc((size_t)BGR * CH0 * 2);
  bf16*  sbf    = (bf16*)alloc((size_t)BGR * CH0 * 2);
  float* out_acc= (float*)alloc((size_t)BGR * FD * 4);
  float* xrA    = (float*)alloc((size_t)FD * BGR * 4);
  float* xrB    = (float*)alloc((size_t)FD * BGR * 4);
  float* sm     = (float*)alloc((size_t)FD * BGR * 4);
  float* f0     = (float*)alloc((size_t)64 * BGR * 4);
  float* f1     = (float*)alloc((size_t)64 * BGR * 4);
  (void)ws_size; (void)out_size; (void)n_in;

  auto cvt = [&](const float* s, bf16* d, size_t n) {
    k_cvt_bf16<<<dim3((unsigned)((n + 255) / 256)), 256, 0, stream>>>(s, d, n);
  };
  // ---- weight conversions (f32 -> bf16) ----
  cvt(P(0), wb_W011, (size_t)KCAT * CH0);
  cvt(P(1), wb_Wagg, (size_t)CIN * DAGG);
  cvt(P(2), wb_Wl21, (size_t)CH0 * CH0);
  cvt(P(3), wb_Wr21, (size_t)CH0 * CH0);
  for (int e = 0; e < 3; ++e) {
    cvt(P(9 + e * 6 + 0), wb_Wa[e], (size_t)2 * CH0 * CH0);
    cvt(P(9 + e * 6 + 1), wb_Wb[e], (size_t)CH0 * CH0);
  }
  for (int s = 0; s < 3; ++s) {
    cvt(P(161 + s * 3 + 0), wb_sWl[s], (size_t)CH0 * FD);
    cvt(P(161 + s * 3 + 1), wb_sWr[s], (size_t)CH0 * FD);
  }

  // ---- graph segment sums (sorted batch => contiguous per-graph ranges) ----
  k_zero_i32<<<(BGR + 255) / 256, 256, 0, stream>>>(counts, BGR);
  k_count<<<(NN + 255) / 256, 256, 0, stream>>>(batch, counts, NN);
  k_scan<<<1, 32, 0, stream>>>(counts, starts, BGR);
  k_gsum<<<BGR, 256, 0, stream>>>(x, starts, counts, meanL, hcat);

  // ---- emb = meanL @ W_agg + b_agg -> hcat[:, :1200] (bf16) ----
  {
    dim3 g((DAGG + 63) / 64, (BGR + 63) / 64);
    k_gemm<AM_PLAIN, EPI_BF16><<<g, 128, 0, stream>>>(
        meanL, CIN, nullptr, nullptr, nullptr, wb_Wagg, DAGG,
        BGR, DAGG, CIN, CIN, nullptr, hcat, KPAD, P(5), nullptr, nullptr, nullptr, 0);
  }
  // ---- h = relu(g01*(hcat @ W011 + b011) + be01) -> h_bf ----
  {
    dim3 g((CH0 + 63) / 64, (BGR + 63) / 64);
    k_gemm<AM_PLAIN, EPI_GNRELU_BF16><<<g, 128, 0, stream>>>(
        hcat, KPAD, nullptr, nullptr, nullptr, wb_W011, CH0,
        BGR, CH0, KPAD, KCAT, nullptr, h_bf, CH0, P(4), P(27), P(6), nullptr, 0);
  }

  k_fill_f32<<<(BGR * FD + 255) / 256, 256, 0, stream>>>(out_acc, 0.f, BGR * FD);

  // ---- three branches ----
  for (int br = 0; br < 3; ++br) {
    const float* ba = P(9 + br * 6 + 2);
    const float* bb = P(9 + br * 6 + 3);
    const float* be = P(9 + br * 6 + 4);
    const float* gg = P(9 + br * 6 + 5);
    dim3 gE((CH0 + 63) / 64, (E + 63) / 64);
    dim3 gS((CH0 + 63) / 64, (BGR + 63) / 64);
    dim3 gO((FD + 63) / 64, (BGR + 63) / 64);

    // m1 = relu([h_dst, h_src - h_dst] @ Wa + ba)   (edge-gather A mode)
    k_gemm<AM_EDGE, EPI_RELU_BF16><<<gE, 128, 0, stream>>>(
        nullptr, 0, h_bf, esrc, edst, wb_Wa[br], CH0,
        E, CH0, 2 * CH0, 2 * CH0, nullptr, m1, CH0, ba, nullptr, nullptr, nullptr, 0);
    // z = segment_max(m1 @ Wb + bb) over masked edges
    k_fill_f32<<<(BGR * CH0 + 255) / 256, 256, 0, stream>>>(z, -3.402823466e38f, BGR * CH0);
    k_gemm<AM_PLAIN, EPI_AMAX><<<gE, 128, 0, stream>>>(
        m1, CH0, nullptr, nullptr, edst, wb_Wb[br], CH0,
        E, CH0, CH0, CH0, z, nullptr, CH0, bb, nullptr, nullptr, eattr, br);
    // z = relu(g*(finite? z:0) + be)
    k_gbr<<<(BGR * CH0 + 255) / 256, 256, 0, stream>>>(z, z, z_bf, gg, be, BGR * CH0, CH0 - 1);
    // sage layer 1: t1 = mean_nbr(z) @ Wl21 + bl21 + z @ Wr21
    k_band_agg<<<dim3(BGR, CH0 / 256), 256, 0, stream>>>(z, sbf, CH0, br, BGR);
    k_gemm<AM_PLAIN, EPI_F32><<<gS, 128, 0, stream>>>(
        sbf, CH0, nullptr, nullptr, nullptr, wb_Wl21, CH0,
        BGR, CH0, CH0, CH0, t1, nullptr, CH0, P(8), nullptr, nullptr, nullptr, 0);
    k_gemm<AM_PLAIN, EPI_F32ADD><<<gS, 128, 0, stream>>>(
        z_bf, CH0, nullptr, nullptr, nullptr, wb_Wr21, CH0,
        BGR, CH0, CH0, CH0, t1, nullptr, CH0, nullptr, nullptr, nullptr, nullptr, 0);
    // z2 = relu(g21*t1 + be21)
    k_gbr<<<(BGR * CH0 + 255) / 256, 256, 0, stream>>>(t1, z2, z2_bf, P(28), P(7), BGR * CH0, CH0 - 1);
    // sage layer 2: out_acc += mean_nbr(z2) @ Wl + bl + z2 @ Wr
    k_band_agg<<<dim3(BGR, CH0 / 256), 256, 0, stream>>>(z2, sbf, CH0, br, BGR);
    k_gemm<AM_PLAIN, EPI_F32ADD><<<gO, 128, 0, stream>>>(
        sbf, CH0, nullptr, nullptr, nullptr, wb_sWl[br], FD,
        BGR, FD, CH0, CH0, out_acc, nullptr, FD, P(161 + br * 3 + 2), nullptr, nullptr, nullptr, 0);
    k_gemm<AM_PLAIN, EPI_F32ADD><<<gO, 128, 0, stream>>>(
        z2_bf, CH0, nullptr, nullptr, nullptr, wb_sWr[br], FD,
        BGR, FD, CH0, CH0, out_acc, nullptr, FD, nullptr, nullptr, nullptr, nullptr, 0);
  }

  // ---- output slot 0 + xr0 = out.T ----
  k_out0<<<(BGR * FD + 255) / 256, 256, 0, stream>>>(out_acc, dout, xrA, BGR);

  // ---- refinement chain (f32 VALU; ~0.4% of total FLOPs) ----
  float* xr = xrA; float* xrn = xrB;
  for (int r = 0; r < 3; ++r) {
    int rb = 29 + r * 44;
    const float* b_in  = P(rb + 0);
    const float* b_out = P(rb + 1);
    const float* w_in  = P(rb + 42);
    const float* w_out = P(rb + 43);
    k_softmax<<<BGR, 128, 0, stream>>>(xr, sm, BGR);
    k_convin<<<BGR, 64, 0, stream>>>(sm, w_in, b_in, f0, BGR);
    float* fa = f0; float* fb = f1;
    for (int l = 0; l < 10; ++l) {
      int lb = rb + 2 + l * 4;  // {b1, bd, w1, wd}
      k_dil<<<BGR, 64, 0, stream>>>(fa, fb, P(lb + 3), P(lb + 1), P(lb + 2), P(lb + 0),
                                    1 << l, BGR);
      float* tp = fa; fa = fb; fb = tp;
    }
    k_convout<<<BGR, 128, 0, stream>>>(fa, w_out, b_out, xrn,
                                       dout + (size_t)(r + 1) * BGR * FD, BGR);
    float* tp = xr; xr = xrn; xrn = tp;
  }
}
